// GATWithSingleHead_24833500905999
// MI455X (gfx1250) — compile-verified, matrix operand order checked by reading, not run on hardware
//
#include <hip/hip_runtime.h>
#include <hip/hip_bf16.h>
#include <cstddef>

typedef __attribute__((ext_vector_type(16))) __bf16 v16bf;
typedef __attribute__((ext_vector_type(8)))  __bf16 v8bf;
typedef __attribute__((ext_vector_type(4)))  __bf16 v4bf;
typedef __attribute__((ext_vector_type(8)))  float  v8f;
typedef __attribute__((ext_vector_type(4)))  unsigned uint4v;
typedef __attribute__((ext_vector_type(8)))  int      int8v;
typedef __attribute__((ext_vector_type(4)))  int      int4v;

static constexpr int Nn   = 10000;   // nodes
static constexpr int Hh   = 4;       // heads
static constexpr int Ff   = 128;     // features / head
static constexpr int HF   = Hh * Ff; // 512
static constexpr int Ed   = 320000;  // directed edges
static constexpr int ETOT = Ed + Nn; // + self loops
static constexpr float SLOPE = 0.2f;

#if defined(__HIP_DEVICE_COMPILE__) && __has_builtin(__builtin_amdgcn_tensor_load_to_lds)
#define USE_TDM 1
#endif

// ---------------------------------------------------------------------------
// order-preserving float<->uint for atomicMax-based segment max
// ---------------------------------------------------------------------------
__device__ __forceinline__ unsigned f32_order(float f) {
  unsigned u = __float_as_uint(f);
  return (u & 0x80000000u) ? ~u : (u | 0x80000000u);
}
__device__ __forceinline__ float f32_unorder(unsigned u) {
  return __uint_as_float((u & 0x80000000u) ? (u & 0x7fffffffu) : ~u);
}

// ---------------------------------------------------------------------------
// W [K, 512] f32 row-major  ->  Wt [512, K] bf16 row-major
// ---------------------------------------------------------------------------
__global__ void transpose_convert(const float* __restrict__ W,
                                  __bf16* __restrict__ Wt, int K, int J) {
  int idx = blockIdx.x * 256 + threadIdx.x;
  if (idx >= K * J) return;
  int j = idx / K;
  int k = idx - j * K;
  Wt[(size_t)j * K + k] = (__bf16)W[(size_t)k * J + j];
}

// ---------------------------------------------------------------------------
// Tiled bf16 WMMA GEMM: C[M,Nc] = A_f32[M,K] @ Bt_bf16[Nc,K]^T
// block tile 128x128, K-step 32, 8 waves, wave tile 32x64 (2x4 wmma frags).
// B tile is moved by the Tensor Data Mover (TDM) with hardware bank-padding
// (pad every 16 DWORDs by 4 DWORDs -> LDS row stride 40 bf16); A tile is
// staged by the waves with on-the-fly f32->bf16 conversion.
// EPI: 0 = plain f32 store, 1 = sigmoid (decoder)
// ---------------------------------------------------------------------------
template <int EPI>
__global__ __launch_bounds__(256)
void gemm_bf16_wmma(const float* __restrict__ A, const __bf16* __restrict__ Bt,
                    float* __restrict__ C, int M, int Nc, int K) {
  __shared__ __bf16 sA[128 * 40];   // 128 rows (M) x 32 k, stride 40 (bank-pad)
  __shared__ __bf16 sB[128 * 40];   // 128 rows (N) x 32 k, stride 40

  const int tid    = threadIdx.x;
  const int lane   = tid & 31;
  const int wid    = tid >> 5;
  const int blockM = blockIdx.y * 128;
  const int blockN = blockIdx.x * 128;
  const int waveM  = (wid >> 1) * 32;   // 4 wave-rows
  const int waveN  = (wid & 1) * 64;    // 2 wave-cols
  const int l16    = lane & 15;
  const int khalf  = (lane < 16) ? 0 : 8;  // 16-bit A/B fragment K pattern

  v8f acc[2][4] = {};

  const int nk = (K + 31) >> 5;
  for (int kt = 0; kt < nk; ++kt) {
    const int k0 = kt << 5;

#if defined(USE_TDM)
    // ---- B tile via Tensor Data Mover: 2D tile 32(k) x 128(n), bf16,
    //      pad 4 DWORDs every 16 DWORDs => LDS row stride 40 bf16.
    //      OOB zero-fill handles both N- and K-tails (tensor_dim = remaining).
    if (wid == 0) {
      unsigned long long ga =
          (unsigned long long)(const void*)(Bt + (size_t)blockN * K + k0);
      unsigned ldsb = (unsigned)(size_t)(void*)&sB[0];
      unsigned td0 = (unsigned)(K - k0);       // remaining k from tile start
      unsigned td1 = (unsigned)(Nc - blockN);  // remaining n from tile start
      uint4v g0;
      g0[0] = 1u;                                        // count=1, user mode
      g0[1] = ldsb;                                      // lds_addr
      g0[2] = (unsigned)(ga & 0xffffffffu);              // global_addr[31:0]
      g0[3] = (unsigned)((ga >> 32) & 0x01ffffffu) | (2u << 30); // [56:32]|type=2
      int8v g1;
      g1[0] = (int)((1u << 16) |      // data_size = 1 (2 bytes)
                    (1u << 20) |      // pad_enable
                    (3u << 22) |      // pad_interval: every 16 DWORDs
                    (3u << 25));      // pad_amount: 4 DWORDs
      g1[1] = (int)((td0 & 0xffffu) << 16);                       // dim0[15:0]
      g1[2] = (int)((td0 >> 16) | ((td1 & 0xffffu) << 16));       // dim0 hi|dim1 lo
      g1[3] = (int)((td1 >> 16) | (32u << 16));                   // dim1 hi|tile0=32
      g1[4] = 128;                                                // tile1=128,tile2=0
      g1[5] = K;                                                  // dim0_stride lo
      g1[6] = 0;                                                  // stride hi / dim1_stride
      g1[7] = 0;
      int4v gz4 = {0, 0, 0, 0};
      int8v gz8 = {0, 0, 0, 0, 0, 0, 0, 0};
      __builtin_amdgcn_tensor_load_to_lds(g0, g1, gz4, gz4, gz8, 0);
    }
#endif

    // prefetch next A K-slab (speculative, global_prefetch_b8)
    if (kt + 1 < nk) {
      int prow = blockM + (tid >> 1);
      prow = prow < M ? prow : M - 1;
      __builtin_prefetch(A + (size_t)prow * K + (k0 + 32) + (tid & 1) * 16, 0, 1);
    }

    // ---- stage A tile (f32 -> bf16), 128x32, 4 elems/thread/pass ----
    if (blockM + 128 <= M && k0 + 32 <= K) {   // interior fast path (no guards)
#pragma unroll
      for (int p = 0; p < 4; ++p) {
        int g = tid + p * 256;
        int r = g >> 3;
        int c = (g & 7) << 2;
        const float4 v = *(const float4*)(A + (size_t)(blockM + r) * K + k0 + c);
        v4bf bv;
        bv[0] = (__bf16)v.x; bv[1] = (__bf16)v.y;
        bv[2] = (__bf16)v.z; bv[3] = (__bf16)v.w;
        *(v4bf*)&sA[r * 40 + c] = bv;
      }
    } else {
#pragma unroll
      for (int p = 0; p < 4; ++p) {
        int g = tid + p * 256;
        int r = g >> 3;
        int c = (g & 7) << 2;
        int row = blockM + r;
        float f0 = 0.f, f1 = 0.f, f2 = 0.f, f3 = 0.f;
        if (row < M && (k0 + c) < K) {        // K % 4 == 0 for all our shapes
          const float4 v = *(const float4*)(A + (size_t)row * K + k0 + c);
          f0 = v.x; f1 = v.y; f2 = v.z; f3 = v.w;
        }
        v4bf bv;
        bv[0] = (__bf16)f0; bv[1] = (__bf16)f1;
        bv[2] = (__bf16)f2; bv[3] = (__bf16)f3;
        *(v4bf*)&sA[r * 40 + c] = bv;
      }
    }

#if defined(USE_TDM)
    if (wid == 0) __builtin_amdgcn_s_wait_tensorcnt(0);
#else
    // ---- fallback: stage Bt tile (bf16), 128x32, 8 elems/thread/pass ----
#pragma unroll
    for (int p = 0; p < 2; ++p) {
      int g = tid + p * 256;
      int r = g >> 2;
      int c = (g & 3) << 3;
      int col = blockN + r;
      v8bf bv;
#pragma unroll
      for (int i = 0; i < 8; ++i) bv[i] = (__bf16)0.0f;
      if (col < Nc && (k0 + c) < K)           // K % 8 == 0 for all our shapes
        bv = *(const v8bf*)(Bt + (size_t)col * K + k0 + c);
      *(v8bf*)&sB[r * 40 + c] = bv;
    }
#endif

    __syncthreads();

    // ---- fragments: lane<16 -> K{0..7,16..23}, lane>=16 -> K{8..15,24..31}
    v16bf af[2], bfr[4];
#pragma unroll
    for (int fi = 0; fi < 2; ++fi) {
      const __bf16* pa = &sA[(waveM + fi * 16 + l16) * 40 + khalf];
      v8bf lo = *(const v8bf*)pa;
      v8bf hi = *(const v8bf*)(pa + 16);
      af[fi] = __builtin_shufflevector(lo, hi, 0,1,2,3,4,5,6,7,8,9,10,11,12,13,14,15);
    }
#pragma unroll
    for (int bi = 0; bi < 4; ++bi) {
      const __bf16* pb = &sB[(waveN + bi * 16 + l16) * 40 + khalf];
      v8bf lo = *(const v8bf*)pb;
      v8bf hi = *(const v8bf*)(pb + 16);
      bfr[bi] = __builtin_shufflevector(lo, hi, 0,1,2,3,4,5,6,7,8,9,10,11,12,13,14,15);
    }

#pragma unroll
    for (int fi = 0; fi < 2; ++fi)
#pragma unroll
      for (int bi = 0; bi < 4; ++bi)
        acc[fi][bi] = __builtin_amdgcn_wmma_f32_16x16x32_bf16(
            false, af[fi], false, bfr[bi], (short)0, acc[fi][bi], false, false);

    __syncthreads();
  }

  // ---- store: lane<16 covers M rows {0..7}, lane>=16 rows {8..15} of tile
  const int mtop = (lane < 16) ? 0 : 8;
#pragma unroll
  for (int fi = 0; fi < 2; ++fi) {
#pragma unroll
    for (int bi = 0; bi < 4; ++bi) {
      int col = blockN + waveN + bi * 16 + l16;
      if (col < Nc) {
#pragma unroll
        for (int i = 0; i < 8; ++i) {
          int row = blockM + waveM + fi * 16 + mtop + i;
          if (row < M) {
            float v = acc[fi][bi][i];
            if (EPI == 1) v = 1.0f / (1.0f + __expf(-v));
            C[(size_t)row * Nc + col] = v;
          }
        }
      }
    }
  }
}

// ---------------------------------------------------------------------------
// per-(node,head) attention logits: al_s = <h, a_src>, al_d = <h, a_dst>
// one wave per (n, head); 32 lanes x 4 floats reduce F=128
// ---------------------------------------------------------------------------
__global__ __launch_bounds__(256)
void attn_logits(const float* __restrict__ h, const float* __restrict__ a_s,
                 const float* __restrict__ a_d, float* __restrict__ al_s,
                 float* __restrict__ al_d, int total) {
  int w    = blockIdx.x * 8 + (threadIdx.x >> 5);
  int lane = threadIdx.x & 31;
  if (w >= total) return;                 // wave-uniform exit
  int n  = w >> 2;                        // / H
  int hh = w & 3;
  const float* hp  = h  + (size_t)n * HF + hh * Ff + lane * 4;
  const float* asp = a_s + hh * Ff + lane * 4;
  const float* adp = a_d + hh * Ff + lane * 4;
  float ss = 0.f, sd = 0.f;
#pragma unroll
  for (int i = 0; i < 4; ++i) { float hv = hp[i]; ss += hv * asp[i]; sd += hv * adp[i]; }
#pragma unroll
  for (int off = 16; off; off >>= 1) {
    ss += __shfl_xor(ss, off, 32);
    sd += __shfl_xor(sd, off, 32);
  }
  if (lane == 0) { al_s[w] = ss; al_d[w] = sd; }
}

// ---------------------------------------------------------------------------
__global__ void init_buffers(float* __restrict__ agg, unsigned* __restrict__ mu,
                             float* __restrict__ ssum, int nAgg, int nNH) {
  int idx = blockIdx.x * 256 + threadIdx.x;
  if (idx < nAgg) agg[idx] = 0.f;
  if (idx < nNH) { mu[idx] = 0u; ssum[idx] = 0.f; }
}

// e = leaky_relu(al_s[src] + al_d[dst]); segment max via ordered-uint atomicMax
__global__ void edge_pass1(const int* __restrict__ ei, const float* __restrict__ al_s,
                           const float* __restrict__ al_d, float* __restrict__ ebuf,
                           unsigned* __restrict__ mu) {
  int e = blockIdx.x * 256 + threadIdx.x;
  if (e >= ETOT) return;
  int s, d;
  if (e < Ed) { s = ei[e]; d = ei[Ed + e]; } else { s = d = e - Ed; }
#pragma unroll
  for (int hh = 0; hh < Hh; ++hh) {
    float v = al_s[s * Hh + hh] + al_d[d * Hh + hh];
    v = v > 0.f ? v : SLOPE * v;
    ebuf[(size_t)e * Hh + hh] = v;
    atomicMax(&mu[d * Hh + hh], f32_order(v));
  }
}

// ex = exp(e - m[dst]); segment sum
__global__ void edge_pass2(const int* __restrict__ ei, float* __restrict__ ebuf,
                           const unsigned* __restrict__ mu, float* __restrict__ ssum) {
  int e = blockIdx.x * 256 + threadIdx.x;
  if (e >= ETOT) return;
  int d;
  if (e < Ed) { d = ei[Ed + e]; } else { d = e - Ed; }
#pragma unroll
  for (int hh = 0; hh < Hh; ++hh) {
    float ex = __expf(ebuf[(size_t)e * Hh + hh] - f32_unorder(mu[d * Hh + hh]));
    ebuf[(size_t)e * Hh + hh] = ex;
    atomicAdd(&ssum[d * Hh + hh], ex);
  }
}

// agg[dst] += h[src] * alpha ; one wave per edge, 16 floats per lane
__global__ __launch_bounds__(256)
void edge_aggregate(const int* __restrict__ ei, const float* __restrict__ ebuf,
                    const float* __restrict__ ssum, const float* __restrict__ h,
                    float* __restrict__ agg) {
  int w    = blockIdx.x * 8 + (threadIdx.x >> 5);
  int lane = threadIdx.x & 31;
  if (w >= ETOT) return;                  // wave-uniform exit
  int s, d;
  if (w < Ed) { s = ei[w]; d = ei[Ed + w]; } else { s = d = w - Ed; }
  float alpha[Hh];
#pragma unroll
  for (int hh = 0; hh < Hh; ++hh)
    alpha[hh] = ebuf[(size_t)w * Hh + hh] / ssum[d * Hh + hh];
  const float* hs = h + (size_t)s * HF;
  float* ad = agg + (size_t)d * HF;
#pragma unroll
  for (int j = 0; j < 16; ++j) {
    int idx = lane + j * 32;
    atomicAdd(&ad[idx], hs[idx] * alpha[idx >> 7]);
  }
}

// head mean + bias + activation; mode 1 (final layer): tanh, emit z + z_bf16
__global__ void head_mean(const float* __restrict__ agg, const float* __restrict__ b,
                          float* __restrict__ act, float* __restrict__ z_out,
                          __bf16* __restrict__ z_bf, int mode) {
  int idx = blockIdx.x * 256 + threadIdx.x;
  if (idx >= Nn * Ff) return;
  int n = idx >> 7, f = idx & 127;
  const float* p = agg + (size_t)n * HF + f;
  float v = 0.25f * (p[0] + p[128] + p[256] + p[384]) + b[f];
  if (mode == 0) {
    v = v > 0.f ? v : 0.f;
  } else {
    v = tanhf(v);
    z_out[idx] = v;
    z_bf[idx]  = (__bf16)v;
  }
  act[idx] = v;
}

// ---------------------------------------------------------------------------
extern "C" void kernel_launch(void* const* d_in, const int* in_sizes, int n_in,
                              void* d_out, int out_size, void* d_ws, size_t ws_size,
                              hipStream_t stream) {
  (void)in_sizes; (void)n_in; (void)out_size; (void)ws_size;

  const float* x  = (const float*)d_in[0];
  const int*   ei = (const int*)d_in[1];
  const float* Wm[4] = {(const float*)d_in[2],  (const float*)d_in[6],
                        (const float*)d_in[10], (const float*)d_in[14]};
  const float* As[4] = {(const float*)d_in[3],  (const float*)d_in[7],
                        (const float*)d_in[11], (const float*)d_in[15]};
  const float* Ad[4] = {(const float*)d_in[4],  (const float*)d_in[8],
                        (const float*)d_in[12], (const float*)d_in[16]};
  const float* Bv[4] = {(const float*)d_in[5],  (const float*)d_in[9],
                        (const float*)d_in[13], (const float*)d_in[17]};

  float* adj   = (float*)d_out;                 // [N, N]
  float* z_out = adj + (size_t)Nn * Nn;         // [N, F]

  size_t off = 0;
  auto take = [&](size_t bytes) -> void* {
    void* p = (char*)d_ws + off;
    off += (bytes + 255) & ~(size_t)255;
    return p;
  };
  __bf16*   Wt   = (__bf16*)  take((size_t)HF * Nn * 2);   // worst case 512x10000
  float*    h    = (float*)   take((size_t)Nn * HF * 4);
  float*    agg  = (float*)   take((size_t)Nn * HF * 4);
  float*    act  = (float*)   take((size_t)Nn * Ff * 4);
  __bf16*   zb   = (__bf16*)  take((size_t)Nn * Ff * 2);
  float*    als  = (float*)   take((size_t)Nn * Hh * 4);
  float*    ald  = (float*)   take((size_t)Nn * Hh * 4);
  unsigned* mu   = (unsigned*)take((size_t)Nn * Hh * 4);
  float*    ssum = (float*)   take((size_t)Nn * Hh * 4);
  float*    ebuf = (float*)   take((size_t)ETOT * Hh * 4);

  for (int layer = 0; layer < 4; ++layer) {
    const int K = (layer == 0) ? Nn : Ff;
    const float* Ain = (layer == 0) ? x : act;

    // W [K,512] -> Wt [512,K] bf16
    int nWt = HF * K;
    transpose_convert<<<(nWt + 255) / 256, 256, 0, stream>>>(Wm[layer], Wt, K, HF);

    // h = Ain @ W  (WMMA bf16, f32 accum)
    dim3 gg((HF + 127) / 128, (Nn + 127) / 128);
    gemm_bf16_wmma<0><<<gg, 256, 0, stream>>>(Ain, Wt, h, Nn, HF, K);

    // attention
    attn_logits<<<(Nn * Hh + 7) / 8, 256, 0, stream>>>(h, As[layer], Ad[layer],
                                                       als, ald, Nn * Hh);
    init_buffers<<<(Nn * HF + 255) / 256, 256, 0, stream>>>(agg, mu, ssum,
                                                            Nn * HF, Nn * Hh);
    edge_pass1<<<(ETOT + 255) / 256, 256, 0, stream>>>(ei, als, ald, ebuf, mu);
    edge_pass2<<<(ETOT + 255) / 256, 256, 0, stream>>>(ei, ebuf, mu, ssum);
    edge_aggregate<<<(ETOT + 7) / 8, 256, 0, stream>>>(ei, ebuf, ssum, h, agg);

    head_mean<<<(Nn * Ff + 255) / 256, 256, 0, stream>>>(
        agg, Bv[layer], act, z_out, zb, (layer == 3) ? 1 : 0);
  }

  // adj = sigmoid(z @ z^T): A = z (f32), Bt = z (bf16) — WMMA + sigmoid epilogue
  dim3 gd((Nn + 127) / 128, (Nn + 127) / 128);
  gemm_bf16_wmma<1><<<gd, 256, 0, stream>>>(act, zb, adj, Nn, Nn, Ff);
}
